// GptOSSMoE_28638841929805
// MI455X (gfx1250) — matrix operand order repaired and checked by reading, not compiled, not run on hardware
//
#include <hip/hip_runtime.h>
#include <math.h>

// ---------------- problem constants ----------------
#define Hdim    2048
#define TWO_I   4096
#define I_DIM   2048
#define NEXP    16
#define M_TOK   1024
#define TOPK    4
#define TM      32      // tokens per expert tile
#define NCH     128     // output columns per chunk
#define KC      32      // K per WMMA step (bf16)
#define ALPHA_C 1.702f
#define LIMIT_C 7.0f
#define EPS_C   1e-5f

// ---------------- WMMA types ----------------
typedef __attribute__((ext_vector_type(16))) __bf16 v16bf;
typedef __attribute__((ext_vector_type(8)))  float  v8f;
typedef __attribute__((ext_vector_type(2)))  int    v2i;

union FragU  { uint4 u[2]; v16bf v; };
union B16x16 { __bf16 b[16]; uint4 u[2]; };
union B16x8  { __bf16 b[8];  uint4 u; };

// ---------------- async global->LDS copy (gfx1250 path) ----------------
#if defined(__has_builtin)
#if __has_builtin(__builtin_amdgcn_global_load_async_to_lds_b64)
#define HAVE_ASYNC_LDS 1
#endif
#endif

typedef __attribute__((address_space(1))) void* as1_void_p;
typedef __attribute__((address_space(3))) void* as3_void_p;
typedef __attribute__((address_space(1))) v2i*  as1_v2i_p;
typedef __attribute__((address_space(3))) v2i*  as3_v2i_p;

__device__ __forceinline__ void async_cp_b64(const __bf16* g, __bf16* l) {
#if defined(HAVE_ASYNC_LDS)
    __builtin_amdgcn_global_load_async_to_lds_b64(
        (as1_v2i_p)(as1_void_p)(void*)g,
        (as3_v2i_p)(as3_void_p)(void*)l, 0, 0);
#else
    *(uint2*)l = *(const uint2*)g;
#endif
}
__device__ __forceinline__ void wait_async() {
#if defined(HAVE_ASYNC_LDS)
#if __has_builtin(__builtin_amdgcn_s_wait_asynccnt)
    __builtin_amdgcn_s_wait_asynccnt(0);
#else
    asm volatile("s_wait_asynccnt 0x0" ::: "memory");
#endif
#endif
}

// A fragment: 16x32 bf16.  lane<16 -> row=lane, K = 0..15;
// lane>=16 -> row=lane-16, K = 16..31.
__device__ __forceinline__ v16bf load_A(const __bf16* base, int stride,
                                        int row_base, int k0, int lane) {
    const int half = lane >> 4;
    const __bf16* p = base + (size_t)(row_base + (lane & 15)) * stride + k0 + half * 16;
    FragU f;
    f.u[0] = *(const uint4*)(p);        // K 0..7 of this half
    f.u[1] = *(const uint4*)(p + 8);    // K 8..15 of this half
    return f.v;
}
// B fragment: 32x16 bf16. lane<16 -> col=lane, K=0..15; lane>=16 -> col=lane-16, K=16..31.
__device__ __forceinline__ v16bf load_B(const __bf16* base, int stride,
                                        int col_base, int k0, int lane) {
    const int half = lane >> 4;
    const __bf16* p = base + (size_t)(col_base + (lane & 15)) * stride + k0 + half * 16;
    FragU f;
    f.u[0] = *(const uint4*)(p);
    f.u[1] = *(const uint4*)(p + 8);
    return f.v;
}

// convert 16 consecutive f32 -> 16 consecutive bf16 in LDS (native v_cvt path)
__device__ __forceinline__ void cvt16(const float* __restrict__ src, __bf16* dst) {
    B16x16 t;
#pragma unroll
    for (int i = 0; i < 4; ++i) {
        float4 f = ((const float4*)src)[i];
        t.b[4*i+0] = (__bf16)f.x; t.b[4*i+1] = (__bf16)f.y;
        t.b[4*i+2] = (__bf16)f.z; t.b[4*i+3] = (__bf16)f.w;
    }
    ((uint4*)dst)[0] = t.u[0];
    ((uint4*)dst)[1] = t.u[1];
}

// ======================================================================
// Kernel 1: RMSNorm + router (one 256-thread block per token)
// ======================================================================
__global__ __launch_bounds__(256)
void rms_router_kernel(const float* __restrict__ x, const float* __restrict__ norm_w,
                       const float* __restrict__ gate_w, const float* __restrict__ gate_b,
                       __bf16* __restrict__ t_ws, int* __restrict__ counts,
                       int* __restrict__ lists, float* __restrict__ gates) {
    __shared__ float part[NEXP * 256];
    __shared__ float red[256];
    __shared__ float logit[NEXP];

    const int m = blockIdx.x;
    const int tid = threadIdx.x;
    const float* xr = x + (size_t)m * Hdim + tid * 8;

    float xv[8];
    float4 a0 = ((const float4*)xr)[0];
    float4 a1 = ((const float4*)xr)[1];
    xv[0]=a0.x; xv[1]=a0.y; xv[2]=a0.z; xv[3]=a0.w;
    xv[4]=a1.x; xv[5]=a1.y; xv[6]=a1.z; xv[7]=a1.w;

    float ss = 0.f;
#pragma unroll
    for (int j = 0; j < 8; ++j) ss += xv[j] * xv[j];
    red[tid] = ss;
    __syncthreads();
    for (int s = 128; s > 0; s >>= 1) {
        if (tid < s) red[tid] += red[tid + s];
        __syncthreads();
    }
    const float rstd = rsqrtf(red[0] * (1.0f / Hdim) + EPS_C);

    float tv[8];
#pragma unroll
    for (int j = 0; j < 8; ++j) tv[j] = xv[j] * rstd * norm_w[tid * 8 + j];

    // store t as bf16 (one uint4)
    B16x8 pk;
#pragma unroll
    for (int j = 0; j < 8; ++j) pk.b[j] = (__bf16)tv[j];
    *(uint4*)(t_ws + (size_t)m * Hdim + tid * 8) = pk.u;

    // router logits: per-thread partials for all 16 experts
#pragma unroll
    for (int e = 0; e < NEXP; ++e) {
        const float* g = gate_w + (size_t)e * Hdim + tid * 8;
        float p = 0.f;
#pragma unroll
        for (int j = 0; j < 8; ++j) p += tv[j] * g[j];
        part[e * 256 + tid] = p;
    }
    __syncthreads();
    if (tid < NEXP) {
        float s = gate_b[tid];
        for (int i = 0; i < 256; ++i) s += part[tid * 256 + i];
        logit[tid] = s;
    }
    __syncthreads();
    if (tid == 0) {
        float lg[NEXP];
#pragma unroll
        for (int i = 0; i < NEXP; ++i) lg[i] = logit[i];
        int   idx[TOPK];
        float val[TOPK];
#pragma unroll
        for (int k = 0; k < TOPK; ++k) {
            int bi = 0; float bv = lg[0];
            for (int i = 1; i < NEXP; ++i)
                if (lg[i] > bv) { bv = lg[i]; bi = i; }   // strict > : first-index tie-break
            idx[k] = bi; val[k] = bv; lg[bi] = -3.4e38f;
        }
        const float mx = val[0];
        float ex[TOPK], s = 0.f;
#pragma unroll
        for (int k = 0; k < TOPK; ++k) { ex[k] = __expf(val[k] - mx); s += ex[k]; }
        const float inv = 1.0f / s;
#pragma unroll
        for (int k = 0; k < TOPK; ++k) {
            gates[m * TOPK + k] = ex[k] * inv;
            int slot = atomicAdd(&counts[idx[k]], 1);
            lists[idx[k] * M_TOK + slot] = m * TOPK + k;   // assignment id
        }
    }
}

// ======================================================================
// Kernel 2: grouped expert MLP (fused GEMM1 + SwiGLU + GEMM2), bf16 WMMA
// grid = (M_TOK/TM, NEXP), 256 threads (8 waves), double-buffered slabs
// ======================================================================
extern __shared__ char smem_raw[];

#define W_ELEMS (NCH * KC)   // 4096 bf16 per weight-slab buffer
#define T_ELEMS (TM * KC)    // 1024 bf16 per t-slab buffer
#define OFF_WSL (131072)
#define OFF_TSL (131072 + 2 * W_ELEMS * 2)
#define OFF_LST (OFF_TSL + 2 * T_ELEMS * 2)
#define SMEM_SZ (OFF_LST + 256)

__global__ __launch_bounds__(256)
void moe_expert_kernel(const __bf16* __restrict__ t_ws,
                       const float* __restrict__ w1, const float* __restrict__ b1,
                       const float* __restrict__ w2, const float* __restrict__ b2,
                       const int* __restrict__ counts, const int* __restrict__ lists,
                       float* __restrict__ y_ws) {
    __bf16* act  = (__bf16*)smem_raw;                   // 32x2048 bf16 = 128 KB
    __bf16* wsl  = (__bf16*)(smem_raw + OFF_WSL);       // 2 x 128x32 bf16
    __bf16* tsl  = (__bf16*)(smem_raw + OFF_TSL);       // 2 x 32x32 bf16
    int*    slst = (int*)  (smem_raw + OFF_LST);        // 32 ints

    const int e    = blockIdx.y;
    const int cnt  = counts[e];
    const int t0   = blockIdx.x * TM;
    if (t0 >= cnt) return;
    const int valid = min(TM, cnt - t0);

    const int tid  = threadIdx.x;
    const int lane = tid & 31;
    const int wave = tid >> 5;
    const int msub = wave & 1;            // 0..1  (16-row sub-tile)
    const int nb0  = (wave >> 1) * 2;     // n sub-tiles per wave: nb0, nb0+1 (0..7)
    const int nb1  = nb0 + 1;
    const int half = lane >> 4;
    const int col  = lane & 15;

    if (tid < TM) slst[tid] = (tid < valid) ? lists[e * M_TOK + t0 + tid] : 0;
    __syncthreads();

    const float* w1e = w1 + (size_t)e * TWO_I * Hdim;
    const float* w2e = w2 + (size_t)e * Hdim * I_DIM;

    // cooperative-load roles
    const int ln = tid >> 1;   // 0..127 : weight-slab row (output column)
    const int lq = tid & 1;    // 0..1   : which 16-float half of the K slab
    const int tr = tid >> 3;   // 0..31  : t-slab row
    const int tp = tid & 7;    // 0..7   : 4-element part of the K slab
    const __bf16* trow = t_ws + (size_t)(slst[tr] >> 2) * Hdim + tp * 4;

    // -------- GEMM1 (t @ w1^T) + clamped SwiGLU, act -> LDS --------
    for (int ch = 0; ch < TWO_I / NCH; ++ch) {
        v8f c0 = {0.f,0.f,0.f,0.f,0.f,0.f,0.f,0.f};
        v8f c1 = {0.f,0.f,0.f,0.f,0.f,0.f,0.f,0.f};
        const float* w1c = w1e + (size_t)(ch * NCH) * Hdim + (size_t)ln * Hdim + lq * 16;

        __syncthreads();                 // buffers free from previous chunk
        // preload slab 0 into buffer 0
        cvt16(w1c, wsl + ln * KC + lq * 16);
        async_cp_b64(trow, tsl + tr * KC + tp * 4);

        int cur = 0;
        for (int ks = 0; ks < Hdim / KC; ++ks) {
            wait_async();
            __syncthreads();
            if (ks + 1 < Hdim / KC) {
                const int nb = cur ^ 1;
                const int k1 = (ks + 1) * KC;
                cvt16(w1c + k1, wsl + nb * W_ELEMS + ln * KC + lq * 16);
                async_cp_b64(trow + k1, tsl + nb * T_ELEMS + tr * KC + tp * 4);
                __builtin_prefetch(w1c + k1 + KC, 0, 0);
            }
            const __bf16* wb = wsl + cur * W_ELEMS;
            v16bf af = load_A(tsl + cur * T_ELEMS, KC, msub * 16, 0, lane);
            v16bf b0 = load_B(wb, KC, nb0 * 16, 0, lane);
            v16bf b1f= load_B(wb, KC, nb1 * 16, 0, lane);
            c0 = __builtin_amdgcn_wmma_f32_16x16x32_bf16(false, af, false, b0,  (short)0, c0, false, false);
            c1 = __builtin_amdgcn_wmma_f32_16x16x32_bf16(false, af, false, b1f, (short)0, c1, false, false);
            cur ^= 1;
        }
        // bias + clamped SwiGLU (even h-col = gate "a", odd = linear "b")
        auto do_act = [&](v8f c, int nb) {
            const int hcol = ch * NCH + nb * 16 + col;
            const float bias = b1[(size_t)e * TWO_I + hcol];
#pragma unroll
            for (int v = 0; v < 8; ++v) {
                float h = c[v] + bias;
                float other = __shfl_xor(h, 1);
                if (!(lane & 1)) {
                    float aa = fminf(h, LIMIT_C);
                    float bb = fminf(fmaxf(other, -LIMIT_C), LIMIT_C);
                    float g  = aa / (1.0f + __expf(-ALPHA_C * aa));
                    float av = g * (bb + 1.0f);
                    const int row  = msub * 16 + half * 8 + v;
                    const int acol = hcol >> 1;
                    act[row * I_DIM + acol] = (__bf16)av;
                }
            }
        };
        do_act(c0, nb0);
        do_act(c1, nb1);
    }
    __syncthreads();

    // -------- GEMM2 (act @ w2^T) + bias, scatter rows to y_ws --------
    for (int ch = 0; ch < Hdim / NCH; ++ch) {
        v8f d0 = {0.f,0.f,0.f,0.f,0.f,0.f,0.f,0.f};
        v8f d1 = {0.f,0.f,0.f,0.f,0.f,0.f,0.f,0.f};
        const float* w2c = w2e + (size_t)(ch * NCH) * I_DIM + (size_t)ln * I_DIM + lq * 16;

        __syncthreads();
        cvt16(w2c, wsl + ln * KC + lq * 16);

        int cur = 0;
        for (int ks = 0; ks < I_DIM / KC; ++ks) {
            __syncthreads();
            if (ks + 1 < I_DIM / KC) {
                const int k1 = (ks + 1) * KC;
                cvt16(w2c + k1, wsl + (cur ^ 1) * W_ELEMS + ln * KC + lq * 16);
                __builtin_prefetch(w2c + k1 + KC, 0, 0);
            }
            const __bf16* wb = wsl + cur * W_ELEMS;
            v16bf af = load_A(act, I_DIM, msub * 16, ks * KC, lane);  // act already bf16 in LDS
            v16bf b0 = load_B(wb, KC, nb0 * 16, 0, lane);
            v16bf b1f= load_B(wb, KC, nb1 * 16, 0, lane);
            d0 = __builtin_amdgcn_wmma_f32_16x16x32_bf16(false, af, false, b0,  (short)0, d0, false, false);
            d1 = __builtin_amdgcn_wmma_f32_16x16x32_bf16(false, af, false, b1f, (short)0, d1, false, false);
            cur ^= 1;
        }
        auto do_store = [&](v8f d, int nb) {
            const int ocol = ch * NCH + nb * 16 + col;
            const float bias = b2[(size_t)e * Hdim + ocol];
#pragma unroll
            for (int v = 0; v < 8; ++v) {
                const int row = msub * 16 + half * 8 + v;
                if (row < valid) {
                    const int aidx = slst[row];
                    y_ws[(size_t)aidx * Hdim + ocol] = d[v] + bias;
                }
            }
        };
        do_store(d0, nb0);
        do_store(d1, nb1);
    }
}

// ======================================================================
// Kernel 3: combine  out = x + sum_k gate[m,k] * y[m*4+k]
// ======================================================================
__global__ __launch_bounds__(256)
void combine_kernel(const float* __restrict__ x, const float* __restrict__ gates,
                    const float* __restrict__ y_ws, float* __restrict__ out) {
    const int m = blockIdx.x;
    const int d = threadIdx.x * 8;
    const float g0 = gates[m * 4 + 0], g1 = gates[m * 4 + 1];
    const float g2 = gates[m * 4 + 2], g3 = gates[m * 4 + 3];
    const float* xr = x + (size_t)m * Hdim + d;
    const float* y0 = y_ws + (size_t)(m * 4 + 0) * Hdim + d;
    const float* y1 = y_ws + (size_t)(m * 4 + 1) * Hdim + d;
    const float* y2 = y_ws + (size_t)(m * 4 + 2) * Hdim + d;
    const float* y3 = y_ws + (size_t)(m * 4 + 3) * Hdim + d;
    float* o = out + (size_t)m * Hdim + d;
#pragma unroll
    for (int q = 0; q < 2; ++q) {
        float4 xf = ((const float4*)xr)[q];
        float4 a = ((const float4*)y0)[q];
        float4 b = ((const float4*)y1)[q];
        float4 c = ((const float4*)y2)[q];
        float4 e4 = ((const float4*)y3)[q];
        float4 r;
        r.x = xf.x + g0*a.x + g1*b.x + g2*c.x + g3*e4.x;
        r.y = xf.y + g0*a.y + g1*b.y + g2*c.y + g3*e4.y;
        r.z = xf.z + g0*a.z + g1*b.z + g2*c.z + g3*e4.z;
        r.w = xf.w + g0*a.w + g1*b.w + g2*c.w + g3*e4.w;
        ((float4*)o)[q] = r;
    }
}

// ======================================================================
extern "C" void kernel_launch(void* const* d_in, const int* in_sizes, int n_in,
                              void* d_out, int out_size, void* d_ws, size_t ws_size,
                              hipStream_t stream) {
    const float* x      = (const float*)d_in[0];
    const float* norm_w = (const float*)d_in[1];
    const float* gate_w = (const float*)d_in[2];
    const float* gate_b = (const float*)d_in[3];
    const float* w1     = (const float*)d_in[4];
    const float* b1     = (const float*)d_in[5];
    const float* w2     = (const float*)d_in[6];
    const float* b2     = (const float*)d_in[7];
    float* out = (float*)d_out;

    // workspace layout
    char* ws = (char*)d_ws;
    __bf16* t_ws  = (__bf16*)ws;                        // 4 MB   (M x H bf16)
    int*    cnts  = (int*)(ws + 4194304);               // 64 B
    int*    lists = (int*)(ws + 4194368);               // 64 KB  (E x M)
    float*  gates = (float*)(ws + 4259904);             // 16 KB  (M x 4)
    float*  y_ws  = (float*)(ws + 4276288);             // 32 MB  (4M x H)

    (void)hipMemsetAsync(cnts, 0, NEXP * sizeof(int), stream);

    rms_router_kernel<<<M_TOK, 256, 0, stream>>>(x, norm_w, gate_w, gate_b,
                                                 t_ws, cnts, lists, gates);

    (void)hipFuncSetAttribute((const void*)moe_expert_kernel,
                              hipFuncAttributeMaxDynamicSharedMemorySize, (int)SMEM_SZ);
    dim3 gridB(M_TOK / TM, NEXP);
    moe_expert_kernel<<<gridB, 256, SMEM_SZ, stream>>>(t_ws, w1, b1, w2, b2,
                                                       cnts, lists, y_ws);

    combine_kernel<<<M_TOK, 256, 0, stream>>>(x, gates, y_ws, out);
}
